// Dictionary_Learning_61177514164873
// MI455X (gfx1250) — matrix operand order repaired and checked by reading, not compiled, not run on hardware
//
#include <hip/hip_runtime.h>
#include <math.h>

#define Nn 1024
#define Gg 4096
#define Bb 2048
#define MAXT 16

typedef __attribute__((ext_vector_type(2)))  float     v2f;
typedef __attribute__((ext_vector_type(8)))  float     v8f;
typedef __attribute__((ext_vector_type(2)))  _Float16  v2h;
typedef __attribute__((ext_vector_type(4)))  _Float16  v4h;
typedef __attribute__((ext_vector_type(8)))  _Float16  v8h;
typedef __attribute__((ext_vector_type(16))) _Float16  v16h;

// Shared magnitude helper: identical rounding in producer & consumer kernels
// so the float-equality mask test is exact.
__device__ __forceinline__ float mag2(float r, float i) {
    return __fmaf_rn(i, i, r * r);
}

// 2-limb f16 split: x ~= (float)hi + (float)lo, |err| ~ 2^-22 * |x|
struct h2lim { _Float16 hi, lo; };
__device__ __forceinline__ h2lim split2(float x) {
    h2lim r;
    r.hi = (_Float16)x;
    r.lo = (_Float16)(x - (float)r.hi);
    return r;
}

__global__ void init_kernel(int* __restrict__ rowmax, int* __restrict__ cnt) {
    int i = blockIdx.x * blockDim.x + threadIdx.x;
    if (i < Bb) { rowmax[i] = 0; cnt[i] = 0; }
}

// PSI = [a*cos(theta), a*sin(theta)], a = tanh(alpha)/32, split into f16 limbs
// and stored TRANSPOSED as [g][n] (G-major) via an LDS-tiled transpose so both
// the GEMM B-staging and the y-gather read contiguously.
__global__ __launch_bounds__(256)
void psi_split_t_kernel(const float* __restrict__ theta, const float* __restrict__ alpha,
                        _Float16* __restrict__ PhrT, _Float16* __restrict__ PlrT,
                        _Float16* __restrict__ PhiT, _Float16* __restrict__ PliT) {
    __shared__ _Float16 tHr[64][66];
    __shared__ _Float16 tLr[64][66];
    __shared__ _Float16 tHi[64][66];
    __shared__ _Float16 tLi[64][66];
    const int tid = threadIdx.x;
    const int g0 = blockIdx.x * 64;
    const int n0 = blockIdx.y * 64;

    #pragma unroll
    for (int it = 0; it < 16; ++it) {            // load 64(n) x 64(g), g-coalesced
        int idx = tid + it * 256;
        int nl = idx >> 6, gl = idx & 63;
        int gi = (n0 + nl) * Gg + g0 + gl;
        float a = tanhf(alpha[gi]) * (1.0f / 32.0f);
        float s, c;
        sincosf(theta[gi], &s, &c);
        h2lim pr = split2(a * c);
        h2lim pi = split2(a * s);
        tHr[nl][gl] = pr.hi;  tLr[nl][gl] = pr.lo;
        tHi[nl][gl] = pi.hi;  tLi[nl][gl] = pi.lo;
    }
    __syncthreads();
    #pragma unroll
    for (int it = 0; it < 8; ++it) {             // store [g][n], n-coalesced, packed pairs
        int idx = tid + it * 256;
        int gl = idx >> 5, np = (idx & 31) * 2;
        size_t o = (size_t)(g0 + gl) * Nn + n0 + np;
        v2h p;
        p[0] = tHr[np][gl]; p[1] = tHr[np + 1][gl]; *(v2h*)&PhrT[o] = p;
        p[0] = tLr[np][gl]; p[1] = tLr[np + 1][gl]; *(v2h*)&PlrT[o] = p;
        p[0] = tHi[np][gl]; p[1] = tHi[np + 1][gl]; *(v2h*)&PhiT[o] = p;
        p[0] = tLi[np][gl]; p[1] = tLi[np + 1][gl]; *(v2h*)&PliT[o] = p;
    }
}

// Split H into f16 limbs (row-major [b][n], same as input).
__global__ __launch_bounds__(256)
void h_split_kernel(const float* __restrict__ Hr, const float* __restrict__ Hi,
                    _Float16* __restrict__ Ahr, _Float16* __restrict__ Alr,
                    _Float16* __restrict__ Ahi, _Float16* __restrict__ Ali) {
    int i4 = (blockIdx.x * 256 + threadIdx.x) * 4;
    v4h hr, lr, hi, li;
    #pragma unroll
    for (int j = 0; j < 4; ++j) {
        h2lim r = split2(Hr[i4 + j]);
        h2lim m = split2(Hi[i4 + j]);
        hr[j] = r.hi;  lr[j] = r.lo;
        hi[j] = m.hi;  li[j] = m.lo;
    }
    *(v4h*)&Ahr[i4] = hr;  *(v4h*)&Alr[i4] = lr;
    *(v4h*)&Ahi[i4] = hi;  *(v4h*)&Ali[i4] = li;
}

// Complex GEMM x = H @ PSI via 2-limb f16 split on V_WMMA_F32_16X16X32_F16,
// double-buffered in LDS: chunk k+1's global loads fly under chunk k's WMMAs.
// Per real product: Ah*Bh + Ah*Bl + Al*Bh (Al*Bl dropped, ~2^-22 error).
// Workgroup = 8 waves -> 32(B) x 64(G); K staged 32-deep per buffer.
__global__ __launch_bounds__(256)
void cgemm_wmma_f16x2_kernel(const _Float16* __restrict__ Ahr, const _Float16* __restrict__ Alr,
                             const _Float16* __restrict__ Ahi, const _Float16* __restrict__ Ali,
                             const _Float16* __restrict__ BhrT, const _Float16* __restrict__ BlrT,
                             const _Float16* __restrict__ BhiT, const _Float16* __restrict__ BliT,
                             float* __restrict__ Xr, float* __restrict__ Xi,
                             int* __restrict__ rowmax) {
    // K-chunk = 32 halves; row stride 48 halves (96 B: 32B-aligned, padded); x2 buffers
    __shared__ __align__(64) _Float16 sAhr[2][32][48];
    __shared__ __align__(64) _Float16 sAlr[2][32][48];
    __shared__ __align__(64) _Float16 sAhi[2][32][48];
    __shared__ __align__(64) _Float16 sAli[2][32][48];
    __shared__ __align__(64) _Float16 sBhr[2][64][48];   // transposed: [g][k]
    __shared__ __align__(64) _Float16 sBlr[2][64][48];
    __shared__ __align__(64) _Float16 sBhi[2][64][48];
    __shared__ __align__(64) _Float16 sBli[2][64][48];

    const int tid  = threadIdx.x;
    const int lane = tid & 31;
    const int wave = tid >> 5;
    const int wm   = wave >> 2;               // 0..1 (B dim)
    const int wn   = wave & 3;                // 0..3 (G dim)
    const int row0 = blockIdx.y * 32;
    const int col0 = blockIdx.x * 64;

    v8f acc_rr = {}, acc_ii = {}, acc_ri = {}, acc_ir = {};

    const int h    = lane >> 4;               // half-wave group
    const int arow = (wm << 4) + (lane & 15); // LDS H row
    const int bcol = (wn << 4) + (lane & 15); // LDS PSI row (transposed)

    // staging decomposition: A 32x32 halves -> one v4h/thread/array,
    //                        B 64x32 halves -> one v8h/thread/array
    const int ar = tid >> 3, ak = (tid & 7) * 4;
    const int bg = tid >> 2, bk = (tid & 3) * 8;

    v4h rAhr, rAlr, rAhi, rAli;
    v8h rBhr, rBlr, rBhi, rBli;

    auto load_regs = [&](int kc) {
        size_t ga = (size_t)(row0 + ar) * Nn + kc + ak;
        rAhr = *(const v4h*)&Ahr[ga];
        rAlr = *(const v4h*)&Alr[ga];
        rAhi = *(const v4h*)&Ahi[ga];
        rAli = *(const v4h*)&Ali[ga];
        size_t gb = (size_t)(col0 + bg) * Nn + kc + bk;
        rBhr = *(const v8h*)&BhrT[gb];
        rBlr = *(const v8h*)&BlrT[gb];
        rBhi = *(const v8h*)&BhiT[gb];
        rBli = *(const v8h*)&BliT[gb];
    };
    auto store_lds = [&](int buf) {
        *(v4h*)&sAhr[buf][ar][ak] = rAhr;
        *(v4h*)&sAlr[buf][ar][ak] = rAlr;
        *(v4h*)&sAhi[buf][ar][ak] = rAhi;
        *(v4h*)&sAli[buf][ar][ak] = rAli;
        *(v8h*)&sBhr[buf][bg][bk] = rBhr;
        *(v8h*)&sBlr[buf][bg][bk] = rBlr;
        *(v8h*)&sBhi[buf][bg][bk] = rBhi;
        *(v8h*)&sBli[buf][bg][bk] = rBli;
    };

    // prologue: fill buffer 0
    load_regs(0);
    store_lds(0);
    __syncthreads();

    int buf = 0;
    for (int kc = 0; kc < Nn; kc += 32) {
        const bool more = (kc + 32) < Nn;
        if (more) load_regs(kc + 32);         // global loads in flight under WMMAs

        // A fragment: halves [16h..16h+15] (contiguous, matches B)
        v8h t0, t1;
        t0 = *(const v8h*)&sAhr[buf][arow][16 * h];
        t1 = *(const v8h*)&sAhr[buf][arow][16 * h + 8];
        v16h fAhr = __builtin_shufflevector(t0, t1, 0,1,2,3,4,5,6,7,8,9,10,11,12,13,14,15);
        t0 = *(const v8h*)&sAlr[buf][arow][16 * h];
        t1 = *(const v8h*)&sAlr[buf][arow][16 * h + 8];
        v16h fAlr = __builtin_shufflevector(t0, t1, 0,1,2,3,4,5,6,7,8,9,10,11,12,13,14,15);
        t0 = *(const v8h*)&sAhi[buf][arow][16 * h];
        t1 = *(const v8h*)&sAhi[buf][arow][16 * h + 8];
        v16h fAhi = __builtin_shufflevector(t0, t1, 0,1,2,3,4,5,6,7,8,9,10,11,12,13,14,15);
        t0 = *(const v8h*)&sAli[buf][arow][16 * h];
        t1 = *(const v8h*)&sAli[buf][arow][16 * h + 8];
        v16h fAli = __builtin_shufflevector(t0, t1, 0,1,2,3,4,5,6,7,8,9,10,11,12,13,14,15);
        // B fragment: halves [16h..16h+15] of the K-transposed row (contiguous k order, same as A)
        v16h fBhr = *(const v16h*)&sBhr[buf][bcol][16 * h];
        v16h fBlr = *(const v16h*)&sBlr[buf][bcol][16 * h];
        v16h fBhi = *(const v16h*)&sBhi[buf][bcol][16 * h];
        v16h fBli = *(const v16h*)&sBli[buf][bcol][16 * h];

        // Hr*Pr
        acc_rr = __builtin_amdgcn_wmma_f32_16x16x32_f16(false, fAhr, false, fBhr, (short)0, acc_rr, false, false);
        acc_rr = __builtin_amdgcn_wmma_f32_16x16x32_f16(false, fAhr, false, fBlr, (short)0, acc_rr, false, false);
        acc_rr = __builtin_amdgcn_wmma_f32_16x16x32_f16(false, fAlr, false, fBhr, (short)0, acc_rr, false, false);
        // Hi*Pi
        acc_ii = __builtin_amdgcn_wmma_f32_16x16x32_f16(false, fAhi, false, fBhi, (short)0, acc_ii, false, false);
        acc_ii = __builtin_amdgcn_wmma_f32_16x16x32_f16(false, fAhi, false, fBli, (short)0, acc_ii, false, false);
        acc_ii = __builtin_amdgcn_wmma_f32_16x16x32_f16(false, fAli, false, fBhi, (short)0, acc_ii, false, false);
        // Hr*Pi
        acc_ri = __builtin_amdgcn_wmma_f32_16x16x32_f16(false, fAhr, false, fBhi, (short)0, acc_ri, false, false);
        acc_ri = __builtin_amdgcn_wmma_f32_16x16x32_f16(false, fAhr, false, fBli, (short)0, acc_ri, false, false);
        acc_ri = __builtin_amdgcn_wmma_f32_16x16x32_f16(false, fAlr, false, fBhi, (short)0, acc_ri, false, false);
        // Hi*Pr
        acc_ir = __builtin_amdgcn_wmma_f32_16x16x32_f16(false, fAhi, false, fBhr, (short)0, acc_ir, false, false);
        acc_ir = __builtin_amdgcn_wmma_f32_16x16x32_f16(false, fAhi, false, fBlr, (short)0, acc_ir, false, false);
        acc_ir = __builtin_amdgcn_wmma_f32_16x16x32_f16(false, fAli, false, fBhr, (short)0, acc_ir, false, false);

        if (more) store_lds(buf ^ 1);         // s_wait_loadcnt lands after the WMMAs
        __syncthreads();                      // s_wait_dscnt 0 + barrier: ping-pong safe
        buf ^= 1;
    }

    // Epilogue: combine complex parts, store unmasked x, track per-row max |x|^2.
    const int colg = col0 + bcol;
    #pragma unroll
    for (int v = 0; v < 8; ++v) {
        int rowg = row0 + (wm << 4) + v + (h << 3);
        float xr = acc_rr[v] - acc_ii[v];
        float xi = acc_ri[v] + acc_ir[v];
        Xr[rowg * Gg + colg] = xr;
        Xi[rowg * Gg + colg] = xi;
        float m = mag2(xr, xi);                       // >= 0
        m = fmaxf(m, __shfl_xor(m, 1, 32));
        m = fmaxf(m, __shfl_xor(m, 2, 32));
        m = fmaxf(m, __shfl_xor(m, 4, 32));
        m = fmaxf(m, __shfl_xor(m, 8, 32));           // reduce across the row's 16 cols
        if ((lane & 15) == 0)
            atomicMax(&rowmax[rowg], __float_as_int(m));  // nonneg float: int order == float order
    }
}

// Apply hard top-1 mask in place; record tie positions per row.
__global__ __launch_bounds__(256)
void mask_kernel(float* __restrict__ Xr, float* __restrict__ Xi,
                 const int* __restrict__ rowmax, int* __restrict__ cnt,
                 int* __restrict__ ties) {
    int idx = blockIdx.x * 256 + threadIdx.x;  // over B*G
    int b = idx >> 12;                         // /4096
    int g = idx & 4095;
    float xr = Xr[idx], xi = Xi[idx];
    float m  = mag2(xr, xi);
    float mx = __int_as_float(rowmax[b]);
    if (m == mx) {
        int p = atomicAdd(&cnt[b], 1);
        if (p < MAXT) ties[b * MAXT + p] = g;
        // keep xr/xi (mask == 1)
    } else {
        Xr[idx] = 0.0f;
        Xi[idx] = 0.0f;
    }
}

// y = x @ PSI^H with x row-sparse: gather the selected PSI column(s).
// PSI reconstructed as hi+lo == exactly the value the GEMM consumed.
// y_r = x_r*P_r + x_i*P_i ;  y_i = x_i*P_r - x_r*P_i
__global__ __launch_bounds__(256)
void y_kernel(const float* __restrict__ Xr, const float* __restrict__ Xi,
              const _Float16* __restrict__ PhrT, const _Float16* __restrict__ PlrT,
              const _Float16* __restrict__ PhiT, const _Float16* __restrict__ PliT,
              const int* __restrict__ cnt, const int* __restrict__ ties,
              float* __restrict__ Yr, float* __restrict__ Yi) {
    int b = blockIdx.x;
    int t = threadIdx.x;
    float yr[4] = {0.f, 0.f, 0.f, 0.f};
    float yi[4] = {0.f, 0.f, 0.f, 0.f};
    int nt = cnt[b];
    if (nt > MAXT) nt = MAXT;
    for (int e = 0; e < nt; ++e) {
        int g = ties[b * MAXT + e];
        float xr = Xr[b * Gg + g];
        float xi = Xi[b * Gg + g];
        size_t base = (size_t)g * Nn;
        #pragma unroll
        for (int j = 0; j < 4; ++j) {
            int n = t + j * 256;                          // [g][n]: n-coalesced
            float pr = (float)PhrT[base + n] + (float)PlrT[base + n];
            float pi = (float)PhiT[base + n] + (float)PliT[base + n];
            yr[j] += xr * pr + xi * pi;
            yi[j] += xi * pr - xr * pi;
        }
    }
    #pragma unroll
    for (int j = 0; j < 4; ++j) {
        int n = t + j * 256;
        Yr[b * Nn + n] = yr[j];
        Yi[b * Nn + n] = yi[j];
    }
}

extern "C" void kernel_launch(void* const* d_in, const int* in_sizes, int n_in,
                              void* d_out, int out_size, void* d_ws, size_t ws_size,
                              hipStream_t stream) {
    const float* H     = (const float*)d_in[0];   // [2, B, N]
    const float* theta = (const float*)d_in[1];   // [N, G]
    const float* alpha = (const float*)d_in[2];   // [N, G]
    const float* Hr = H;
    const float* Hi = H + (size_t)Bb * Nn;

    float* out = (float*)d_out;                   // x [2,B,G] then y [2,B,N]
    float* Xr = out;
    float* Xi = out + (size_t)Bb * Gg;
    float* Yr = out + 2ull * Bb * Gg;
    float* Yi = Yr + (size_t)Bb * Nn;

    // Workspace: PSI limbs transposed [G][N] (4 x 8 MB), H limbs [B][N] (4 x 4 MB)
    _Float16* PhrT = (_Float16*)d_ws;
    _Float16* PlrT = PhrT + (size_t)Nn * Gg;
    _Float16* PhiT = PlrT + (size_t)Nn * Gg;
    _Float16* PliT = PhiT + (size_t)Nn * Gg;
    _Float16* Ahr  = PliT + (size_t)Nn * Gg;
    _Float16* Alr  = Ahr + (size_t)Bb * Nn;
    _Float16* Ahi  = Alr + (size_t)Bb * Nn;
    _Float16* Ali  = Ahi + (size_t)Bb * Nn;
    int* rowmax = (int*)(Ali + (size_t)Bb * Nn);
    int* cnt    = rowmax + Bb;
    int* ties   = cnt + Bb;

    init_kernel<<<(Bb + 255) / 256, 256, 0, stream>>>(rowmax, cnt);
    psi_split_t_kernel<<<dim3(Gg / 64, Nn / 64), 256, 0, stream>>>(theta, alpha, PhrT, PlrT, PhiT, PliT);
    h_split_kernel<<<(Bb * Nn) / (256 * 4), 256, 0, stream>>>(Hr, Hi, Ahr, Alr, Ahi, Ali);
    cgemm_wmma_f16x2_kernel<<<dim3(Gg / 64, Bb / 32), 256, 0, stream>>>(
        Ahr, Alr, Ahi, Ali, PhrT, PlrT, PhiT, PliT, Xr, Xi, rowmax);
    mask_kernel<<<(Bb * Gg) / 256, 256, 0, stream>>>(Xr, Xi, rowmax, cnt, ties);
    y_kernel<<<Bb, 256, 0, stream>>>(Xr, Xi, PhrT, PlrT, PhiT, PliT, cnt, ties, Yr, Yi);
}
